// C_GAN_NET_9320079032867
// MI455X (gfx1250) — matrix operand order repaired and checked by reading, not compiled, not run on hardware
//
#include <hip/hip_runtime.h>

// MI455X / gfx1250: wave32, WMMA bf16 16x16x32, f32 accumulate.
// Collapsed C-GAN generator: no temporal recurrence (state slot 0 stays zero,
// slot 1 is overwritten before being read) -> 16384 independent rows through
// 5 GEMMs (16384 x 2048 x 512) with fused LSTM gate nonlinearities.
// M-tile = 32 rows/block: each B-fragment feeds 2 WMMAs (compute-bound regime).

typedef __attribute__((ext_vector_type(16))) __bf16 v16bf;
typedef __attribute__((ext_vector_type(8)))  __bf16 v8bf;
typedef __attribute__((ext_vector_type(8)))  float  v8f;

#define KDIM 512
#define NG   2048
#define MT   32
#define LDST 520          // padded LDS row stride in halves (spreads banks)
#define NROWS (128 * 128) // B*T

__device__ __forceinline__ unsigned short f2bf(float f) {
  union { float f; unsigned u; } x; x.f = f;
  unsigned r = (x.u + 0x7FFFu + ((x.u >> 16) & 1u)) >> 16;  // RNE
  return (unsigned short)r;
}
__device__ __forceinline__ float sigmoidf(float x) {
  return 1.0f / (1.0f + __expf(-x));
}

// A-fragment (16x32 bf16) from LDS, contiguous k order (same as B):
// lane m (0..15) / m+16: K = 16*kh .. 16*kh+15 (contiguous)
__device__ __forceinline__ v16bf afrag_lds(const unsigned short* lds, int m, int k0, int kh) {
  union { v16bf v; v8bf h[2]; } u;
  const unsigned short* p = lds + m * LDST + k0 + 16 * kh;
  u.h[0] = *reinterpret_cast<const v8bf*>(p);
  u.h[1] = *reinterpret_cast<const v8bf*>(p + 8);
  return u.v;
}

// 32x64 strip (2 m-frags x 4 n-tiles) accumulated over K=512.
// A from LDS tile (rows 0..31); B = row-major bf16 weight [2048,512]:
// B-frag lane n holds column n, K = lane-half*16 .. +15 (contiguous 32B).
// Each B-fragment is reused by both m-fragments.
__device__ __forceinline__ void gemm_strip(const unsigned short* lsA,
                                           const unsigned short* __restrict__ W,
                                           int nbase, int lane15, int kh,
                                           v8f acc[2][4]) {
#pragma unroll 2
  for (int k0 = 0; k0 < KDIM; k0 += 32) {
    v16bf a0 = afrag_lds(lsA, lane15, k0, kh);
    v16bf a1 = afrag_lds(lsA, lane15 + 16, k0, kh);
#pragma unroll
    for (int t = 0; t < 4; ++t) {
      const unsigned short* bp = W + (size_t)(nbase + t * 16 + lane15) * KDIM + k0 + kh * 16;
      v16bf b = *reinterpret_cast<const v16bf*>(bp);
      acc[0][t] = __builtin_amdgcn_wmma_f32_16x16x32_bf16(false, a0, false, b, (short)0,
                                                          acc[0][t], false, false);
      acc[1][t] = __builtin_amdgcn_wmma_f32_16x16x32_bf16(false, a1, false, b, (short)0,
                                                          acc[1][t], false, false);
    }
  }
}

__device__ __forceinline__ void zero8(v8f acc[2][4]) {
#pragma unroll
  for (int m = 0; m < 2; ++m)
#pragma unroll
    for (int t = 0; t < 4; ++t) {
      v8f z = {0.f, 0.f, 0.f, 0.f, 0.f, 0.f, 0.f, 0.f};
      acc[m][t] = z;
    }
}

// Stage 32 rows x 512 bf16 into padded LDS (u32 copies).
__device__ __forceinline__ void stage32(const unsigned short* __restrict__ src,
                                        unsigned short* lds, int m0) {
  const unsigned* s = reinterpret_cast<const unsigned*>(src + (size_t)m0 * KDIM);
  unsigned* d = reinterpret_cast<unsigned*>(lds);
  for (int i = threadIdx.x; i < MT * (KDIM / 2); i += 256) {
    int row = i >> 8;       // 256 u32 per source row
    int c = i & 255;
    d[row * (LDST / 2) + c] = s[row * (KDIM / 2) + c];
  }
}

__global__ void cvt_kernel(const float* __restrict__ in, unsigned short* __restrict__ out, int n) {
  int i = blockIdx.x * 256 + threadIdx.x;
  if (i < n) out[i] = f2bf(in[i]);
}

__global__ void addbias_kernel(const float* __restrict__ a, const float* __restrict__ b,
                               float* __restrict__ out, int n) {
  int i = blockIdx.x * 256 + threadIdx.x;
  if (i < n) out[i] = a[i] + b[i];
}

// Fused LSTM cell: gates = X*Wih^T (+ Hprev*Whh^T) + bias; gate order i,f,g,o.
// Wave w owns h-columns [w*64, w*64+64); sections processed g -> i -> f -> o
// with a running register state so only 8 accumulators + 8 state v8f live.
template <bool LAYER1>
__global__ __launch_bounds__(256) void cell_kernel(
    const unsigned short* __restrict__ xb,     // [16384,512] bf16
    const unsigned short* __restrict__ hprev,  // [16384,512] bf16 (LAYER1)
    const unsigned short* __restrict__ Wih,    // [2048,512]  bf16
    const unsigned short* __restrict__ Whh,    // [2048,512]  bf16 (LAYER1)
    const float* __restrict__ cbias,           // [2048] b_ih+b_hh
    const float* __restrict__ c_in,            // [16384,512] f32 (LAYER1)
    unsigned short* __restrict__ h_out,        // [16384,512] bf16
    float* __restrict__ c_out) {               // [16384,512] f32 (!LAYER1)
  __shared__ unsigned short sX[MT * LDST];
  __shared__ unsigned short sH[MT * LDST];

  const int m0 = blockIdx.x * MT;
  stage32(xb, sX, m0);
  if (LAYER1) stage32(hprev, sH, m0);
  __syncthreads();

  const int lane = threadIdx.x & 31;
  const int w = threadIdx.x >> 5;
  const int lane15 = lane & 15;
  const int kh = lane >> 4;
  const int cb = w * 64;   // this wave's h-column base

  float st[2][4][8];       // running elementwise state (c candidate)
  v8f acc[2][4];

  // ---- g gate (gates cols 1024+cb) ----
  zero8(acc);
  gemm_strip(sX, Wih, 1024 + cb, lane15, kh, acc);
  if (LAYER1) gemm_strip(sH, Whh, 1024 + cb, lane15, kh, acc);
#pragma unroll
  for (int m = 0; m < 2; ++m)
#pragma unroll
    for (int t = 0; t < 4; ++t) {
      float b = cbias[1024 + cb + t * 16 + lane15];
#pragma unroll
      for (int r = 0; r < 8; ++r) st[m][t][r] = tanhf(acc[m][t][r] + b);
    }

  // ---- i gate (cols 0+cb): st = sigmoid(i)*tanh(g) ----
  zero8(acc);
  gemm_strip(sX, Wih, cb, lane15, kh, acc);
  if (LAYER1) gemm_strip(sH, Whh, cb, lane15, kh, acc);
#pragma unroll
  for (int m = 0; m < 2; ++m)
#pragma unroll
    for (int t = 0; t < 4; ++t) {
      float b = cbias[cb + t * 16 + lane15];
#pragma unroll
      for (int r = 0; r < 8; ++r) st[m][t][r] *= sigmoidf(acc[m][t][r] + b);
    }

  // ---- f gate: c = sigmoid(f)*c_prev + st. Layer0: c_prev == 0, skip GEMM. ----
  if (LAYER1) {
    zero8(acc);
    gemm_strip(sX, Wih, 512 + cb, lane15, kh, acc);
    gemm_strip(sH, Whh, 512 + cb, lane15, kh, acc);
#pragma unroll
    for (int m = 0; m < 2; ++m)
#pragma unroll
      for (int t = 0; t < 4; ++t) {
        float b = cbias[512 + cb + t * 16 + lane15];
        int col = cb + t * 16 + lane15;
#pragma unroll
        for (int r = 0; r < 8; ++r) {
          int row = m0 + m * 16 + 8 * kh + r;
          float cp = c_in[(size_t)row * KDIM + col];
          st[m][t][r] = sigmoidf(acc[m][t][r] + b) * cp + st[m][t][r];
        }
      }
  } else {
    // st already == c0; persist it for layer 1
#pragma unroll
    for (int m = 0; m < 2; ++m)
#pragma unroll
      for (int t = 0; t < 4; ++t) {
        int col = cb + t * 16 + lane15;
#pragma unroll
        for (int r = 0; r < 8; ++r) {
          int row = m0 + m * 16 + 8 * kh + r;
          c_out[(size_t)row * KDIM + col] = st[m][t][r];
        }
      }
  }

  // ---- o gate: h = sigmoid(o)*tanh(c) ----
  zero8(acc);
  gemm_strip(sX, Wih, 1536 + cb, lane15, kh, acc);
  if (LAYER1) gemm_strip(sH, Whh, 1536 + cb, lane15, kh, acc);
#pragma unroll
  for (int m = 0; m < 2; ++m)
#pragma unroll
    for (int t = 0; t < 4; ++t) {
      float b = cbias[1536 + cb + t * 16 + lane15];
      int col = cb + t * 16 + lane15;
#pragma unroll
      for (int r = 0; r < 8; ++r) {
        int row = m0 + m * 16 + 8 * kh + r;
        float h = sigmoidf(acc[m][t][r] + b) * tanhf(st[m][t][r]);
        h_out[(size_t)row * KDIM + col] = f2bf(h);
      }
    }
}

// out = sigmoid(h @ Whh^T + b_hh), written into d_out[:, out_col0 : out_col0+2048]
__global__ __launch_bounds__(256) void outgate_kernel(
    const unsigned short* __restrict__ hb,   // [16384,512] bf16
    const unsigned short* __restrict__ Whh,  // [2048,512]  bf16
    const float* __restrict__ bias,          // [2048] b_hh
    float* __restrict__ out, int out_col0) {
  __shared__ unsigned short sA[MT * LDST];
  const int m0 = blockIdx.x * MT;
  stage32(hb, sA, m0);
  __syncthreads();

  const int lane = threadIdx.x & 31;
  const int w = threadIdx.x >> 5;
  const int lane15 = lane & 15;
  const int kh = lane >> 4;

  for (int p = 0; p < 4; ++p) {
    int nbase = p * 512 + w * 64;
    v8f acc[2][4];
    zero8(acc);
    gemm_strip(sA, Whh, nbase, lane15, kh, acc);
#pragma unroll
    for (int m = 0; m < 2; ++m)
#pragma unroll
      for (int t = 0; t < 4; ++t) {
        int col = nbase + t * 16 + lane15;
        float b = bias[col];
#pragma unroll
        for (int r = 0; r < 8; ++r) {
          int row = m0 + m * 16 + 8 * kh + r;
          out[(size_t)row * 4096 + out_col0 + col] = sigmoidf(acc[m][t][r] + b);
        }
      }
  }
}

extern "C" void kernel_launch(void* const* d_in, const int* in_sizes, int n_in,
                              void* d_out, int out_size, void* d_ws, size_t ws_size,
                              hipStream_t stream) {
  (void)in_sizes; (void)n_in; (void)out_size; (void)ws_size;
  const float* x    = (const float*)d_in[0];  // [128,128,512]
  const float* W_ih = (const float*)d_in[1];  // [2,2048,512]
  const float* W_hh = (const float*)d_in[2];  // [2,2048,512]
  const float* b_ih = (const float*)d_in[3];  // [2,2048]
  const float* b_hh = (const float*)d_in[4];  // [2,2048]
  float* out = (float*)d_out;                 // [128,128,4096]

  char* ws = (char*)d_ws;
  // workspace layout (bytes)
  unsigned short* xb   = (unsigned short*)(ws);                         // 16,777,216
  unsigned short* wihb = (unsigned short*)(ws + 16777216);              //  4,194,304
  unsigned short* whhb = (unsigned short*)(ws + 20971520);              //  4,194,304
  unsigned short* h0b  = (unsigned short*)(ws + 25165824);              // 16,777,216
  unsigned short* h1b  = (unsigned short*)(ws + 41943040);              // 16,777,216
  float*          c0   = (float*)(ws + 58720256);                       // 33,554,432
  float*          cbias= (float*)(ws + 92274688);                       //     16,384

  int n;
  n = NROWS * KDIM;
  cvt_kernel<<<(n + 255) / 256, 256, 0, stream>>>(x, xb, n);
  n = 2 * NG * KDIM;
  cvt_kernel<<<(n + 255) / 256, 256, 0, stream>>>(W_ih, wihb, n);
  cvt_kernel<<<(n + 255) / 256, 256, 0, stream>>>(W_hh, whhb, n);
  n = 2 * NG;
  addbias_kernel<<<(n + 255) / 256, 256, 0, stream>>>(b_ih, b_hh, cbias, n);

  dim3 grid(NROWS / MT), blk(256);
  // layer 0 cell (h=c=0): gates0 = X*Wih0^T
  cell_kernel<false><<<grid, blk, 0, stream>>>(xb, nullptr, wihb, nullptr,
                                               cbias, nullptr, h0b, c0);
  // out0 = sigmoid(h0*Whh0^T + b_hh0)
  outgate_kernel<<<grid, blk, 0, stream>>>(h0b, whhb, b_hh, out, 0);
  // layer 1 cell: gates1 = X*Wih1^T + h0*Whh1^T, c_prev = c0
  cell_kernel<true><<<grid, blk, 0, stream>>>(xb, h0b,
                                              wihb + (size_t)NG * KDIM,
                                              whhb + (size_t)NG * KDIM,
                                              cbias + NG, c0, h1b, nullptr);
  // out1 = sigmoid(h1*Whh1^T + b_hh1)
  outgate_kernel<<<grid, blk, 0, stream>>>(h1b, whhb + (size_t)NG * KDIM,
                                           b_hh + NG, out, NG);
}
